// MultiHeadAttention_59158879535767
// MI455X (gfx1250) — compile-verified, matrix operand order checked by reading, not run on hardware
//
#include <hip/hip_runtime.h>

typedef __bf16 bf16;
typedef __attribute__((ext_vector_type(4)))  float        f32x4;
typedef __attribute__((ext_vector_type(4)))  bf16         bf16x4;
typedef __attribute__((ext_vector_type(8)))  bf16         bf16x8;
typedef __attribute__((ext_vector_type(16))) bf16         v16bf;
typedef __attribute__((ext_vector_type(8)))  float        v8f;
typedef __attribute__((ext_vector_type(4)))  unsigned int u32x4;
typedef __attribute__((ext_vector_type(4)))  int          i32x4;
typedef __attribute__((ext_vector_type(8)))  int          i32x8;

#define TM 128
#define TN 128
#define TK 64
#define LDP 72   // padded LDS row stride (bf16): 64 data + 8 pad = 144B = 128B + 16B

template <typename OT> __device__ __forceinline__ OT from_f32(float x);
template <> __device__ __forceinline__ float from_f32<float>(float x) { return x; }
template <> __device__ __forceinline__ bf16  from_f32<bf16>(float x)  { return (bf16)x; }

// ---------------------------------------------------------------------------
// TDM: DMA one 128-row x 64-col bf16 tile (row stride 1024 elems in memory)
// into LDS with a 16B pad appended after every 128B row (-> 144B LDS stride).
// D# group1: data_size=1(2B), pad_enable, pad_interval code 4 (32 DWORDs),
// pad_amount code 3 (4 DWORDs). Tracked on TENSORcnt of the issuing wave.
// ---------------------------------------------------------------------------
__device__ __forceinline__ void tdm_load_tile(unsigned lds_bytes, const void* gaddr)
{
  const unsigned long long ga = (unsigned long long)(size_t)gaddr;
  u32x4 g0;
  g0[0] = 1u;                                                // count=1, user D#
  g0[1] = lds_bytes;                                         // lds_addr (bytes)
  g0[2] = (unsigned)ga;                                      // global_addr[31:0]
  g0[3] = (unsigned)((ga >> 32) & 0x01FFFFFFu) | (2u << 30); // [56:32] | type=2
  i32x8 g1;
  g1[0] = (int)((1u << 16) | (1u << 20) | (4u << 22) | (3u << 25));
  g1[1] = (int)(1024u << 16);      // tensor_dim0 = 1024 (bits 79:48, low half)
  g1[2] = (int)(1024u << 16);      // dim0 hi = 0 | tensor_dim1 = 1024 (low 16)
  g1[3] = (int)(64u << 16);        // dim1 hi = 0 | tile_dim0 = 64
  g1[4] = (int)128u;               // tile_dim1 = 128, tile_dim2 = 0
  g1[5] = (int)1024u;              // tensor_dim0_stride = 1024 (low 32)
  g1[6] = 0;                       // stride hi | tensor_dim1_stride lo
  g1[7] = 0;
  i32x4 z4 = {};
#if __clang_major__ >= 23
  i32x8 z8 = {};
  __builtin_amdgcn_tensor_load_to_lds(g0, g1, z4, z4, z8, 0);
#else
  __builtin_amdgcn_tensor_load_to_lds(g0, g1, z4, z4, 0);
#endif
}

// ---- read one 16x32 bf16 fragment (WMMA A/B layout) from an LDS tile row ----
__device__ __forceinline__ v16bf load_frag(const bf16* rowbase, int kk, int kb) {
  const bf16x8 lo = *(const bf16x8*)(rowbase + kk + kb);
  const bf16x8 hv = *(const bf16x8*)(rowbase + kk + 16 + kb);
  v16bf r;
#pragma unroll
  for (int i = 0; i < 8; ++i) { r[i] = lo[i]; r[i + 8] = hv[i]; }
  return r;
}

// out[m][n] = sum_k A[m][k] * Wt[n][k] + bias[n]   (torch Linear: X @ W.T + b)
// A, Wt are bf16; K = N = 1024 fixed (descriptor hardcodes strides).
template <typename OT>
__global__ void __launch_bounds__(256)
gemm_bias_wmma(const bf16* A, const bf16* Wt,
               const float* __restrict__ bias, OT* __restrict__ out,
               int M, int N, int K)
{
  __shared__ bf16 As[2][TM][LDP];
  __shared__ bf16 Bs[2][TN][LDP];

  const int t    = threadIdx.x;
  const int lane = t & 31;
  const int w    = t >> 5;       // wave 0..7
  const int wr   = w >> 2;       // 0..1  (64 rows each)
  const int wc   = w & 3;        // 0..3  (32 cols each)
  const int l16  = lane & 15;
  const int hi   = lane >> 4;    // 0/1
  const int kb   = hi * 8;
  const bool issuer = (w == 0);  // wave 0 drives the TDM

  const int n0 = blockIdx.x * TN;
  const int m0 = blockIdx.y * TM;

  const unsigned ldsA0 = (unsigned)(size_t)&As[0][0][0];
  const unsigned ldsA1 = (unsigned)(size_t)&As[1][0][0];
  const unsigned ldsB0 = (unsigned)(size_t)&Bs[0][0][0];
  const unsigned ldsB1 = (unsigned)(size_t)&Bs[1][0][0];

  v8f zero = {};
  v8f acc[4][2];
#pragma unroll
  for (int mt = 0; mt < 4; ++mt)
#pragma unroll
    for (int nt = 0; nt < 2; ++nt) acc[mt][nt] = zero;

  // prologue: fill buffer 0
  if (issuer) {
    tdm_load_tile(ldsA0, A  + (size_t)m0 * K);
    tdm_load_tile(ldsB0, Wt + (size_t)n0 * K);
    __builtin_amdgcn_s_wait_tensorcnt(0);
  }
  __syncthreads();

  const int NK = K / TK;
  for (int it = 0; it < NK; ++it) {
    const int cur = it & 1;
    if ((it + 1 < NK) && issuer) {               // prefetch next tile via TDM
      const int ktn = (it + 1) * TK;
      tdm_load_tile(cur ? ldsA0 : ldsA1, A  + (size_t)m0 * K + ktn);
      tdm_load_tile(cur ? ldsB0 : ldsB1, Wt + (size_t)n0 * K + ktn);
    }

#pragma unroll
    for (int kk = 0; kk < TK; kk += 32) {
      v16bf a[4], b[2];
#pragma unroll
      for (int mt = 0; mt < 4; ++mt)
        a[mt] = load_frag(&As[cur][wr * 64 + mt * 16 + l16][0], kk, kb);
#pragma unroll
      for (int nt = 0; nt < 2; ++nt)
        b[nt] = load_frag(&Bs[cur][wc * 32 + nt * 16 + l16][0], kk, kb);
#pragma unroll
      for (int mt = 0; mt < 4; ++mt)
#pragma unroll
        for (int nt = 0; nt < 2; ++nt)
          acc[mt][nt] = __builtin_amdgcn_wmma_f32_16x16x32_bf16(
              false, a[mt], false, b[nt], (short)0, acc[mt][nt], false, false);
    }

    __syncthreads();                              // everyone done with buf cur
    if (issuer) __builtin_amdgcn_s_wait_tensorcnt(0);  // next buf landed
    __syncthreads();                              // publish readiness
  }

  // epilogue: bias + store. C/D layout: lane -> N = l16, VGPR r -> M = r + hi*8
#pragma unroll
  for (int nt = 0; nt < 2; ++nt) {
    const int n = n0 + wc * 32 + nt * 16 + l16;
    const float bv = bias[n];
#pragma unroll
    for (int mt = 0; mt < 4; ++mt) {
#pragma unroll
      for (int r = 0; r < 8; ++r) {
        const int m = m0 + wr * 64 + mt * 16 + hi * 8 + r;
        out[(size_t)m * N + n] = from_f32<OT>(acc[mt][nt][r] + bv);
      }
    }
  }
}

// ---- elementwise f32 -> bf16 (n is a multiple of 1024) ----
__global__ void __launch_bounds__(256)
cvt_f32_bf16(const float* __restrict__ in, bf16* __restrict__ outp)
{
  const size_t i = ((size_t)blockIdx.x * 256 + threadIdx.x) * 4;
  f32x4 v = *(const f32x4*)(in + i);
  bf16x4 b;
#pragma unroll
  for (int j = 0; j < 4; ++j) b[j] = (bf16)v[j];
  *(bf16x4*)(outp + i) = b;
}

// Per-token attention on the (DK=64, H=16) reshaped matrices.
// Q[t] is the 64x16 matrix with element (d,h) at offset d*16+h.
// score = Q K^T / 8 -> softmax rows -> attn = score V.  One block per token.
__global__ void __launch_bounds__(64)
token_attention(const bf16* Q, const bf16* __restrict__ Km,
                const bf16* __restrict__ Vm, bf16* Out)
{
  __shared__ bf16 Ks[64][16];
  __shared__ bf16 Vs[64][16];
  const int tkn = blockIdx.x;
  const int d   = threadIdx.x;        // row 0..63
  const size_t base = (size_t)tkn * 1024;

  float qv[16];
  {
    const bf16x8 q0 = *(const bf16x8*)(Q + base + d * 16);
    const bf16x8 q1 = *(const bf16x8*)(Q + base + d * 16 + 8);
#pragma unroll
    for (int h = 0; h < 8; ++h) { qv[h] = (float)q0[h]; qv[h + 8] = (float)q1[h]; }
  }
  *(bf16x8*)&Ks[d][0] = *(const bf16x8*)(Km + base + d * 16);
  *(bf16x8*)&Ks[d][8] = *(const bf16x8*)(Km + base + d * 16 + 8);
  *(bf16x8*)&Vs[d][0] = *(const bf16x8*)(Vm + base + d * 16);
  *(bf16x8*)&Vs[d][8] = *(const bf16x8*)(Vm + base + d * 16 + 8);
  __syncthreads();

  float s[64];
  float mx = -3.0e38f;
#pragma unroll 4
  for (int e = 0; e < 64; ++e) {
    float a = 0.f;
#pragma unroll
    for (int h = 0; h < 16; ++h) a += qv[h] * (float)Ks[e][h];
    a *= 0.125f;                    // 1/sqrt(DK), DK=64
    s[e] = a;
    mx = fmaxf(mx, a);
  }
  float sum = 0.f;
#pragma unroll 4
  for (int e = 0; e < 64; ++e) { s[e] = __expf(s[e] - mx); sum += s[e]; }
  const float inv = 1.f / sum;

  float o[16];
#pragma unroll
  for (int h = 0; h < 16; ++h) o[h] = 0.f;
#pragma unroll 4
  for (int e = 0; e < 64; ++e) {
    const float p = s[e] * inv;
#pragma unroll
    for (int h = 0; h < 16; ++h) o[h] += p * (float)Vs[e][h];
  }

  bf16x8 r0, r1;
#pragma unroll
  for (int h = 0; h < 8; ++h) { r0[h] = (bf16)o[h]; r1[h] = (bf16)o[h + 8]; }
  *(bf16x8*)(Out + base + d * 16)     = r0;   // Out may alias Q (row d private)
  *(bf16x8*)(Out + base + d * 16 + 8) = r1;
}

extern "C" void kernel_launch(void* const* d_in, const int* in_sizes, int n_in,
                              void* d_out, int out_size, void* d_ws, size_t ws_size,
                              hipStream_t stream)
{
  const float* q   = (const float*)d_in[0];
  const float* k   = (const float*)d_in[1];
  const float* v   = (const float*)d_in[2];
  const float* w_q = (const float*)d_in[3];
  const float* b_q = (const float*)d_in[4];
  const float* w_k = (const float*)d_in[5];
  const float* b_k = (const float*)d_in[6];
  const float* w_v = (const float*)d_in[7];
  const float* b_v = (const float*)d_in[8];
  const float* w_o = (const float*)d_in[9];
  const float* b_o = (const float*)d_in[10];
  float* out = (float*)d_out;

  const int M = 4 * 2048;     // B*S
  const int N = 1024;         // D
  const int K = 1024;         // D
  const size_t MD = (size_t)M * N;   // 8M elems
  const size_t DD = (size_t)N * K;   // 1M elems

  bf16* Qb   = (bf16*)d_ws;          // bf16 copies of inputs
  bf16* Kb   = Qb + MD;
  bf16* Vb   = Kb + MD;
  bf16* Wqb  = Vb + MD;              // bf16 copies of weights
  bf16* Wkb  = Wqb + DD;
  bf16* Wvb  = Wkb + DD;
  bf16* Wob  = Wvb + DD;
  bf16* Qlin = Wob + DD;             // projection outputs
  bf16* Klin = Qlin + MD;
  bf16* Vlin = Klin + MD;

  const dim3 cb(256);
  cvt_f32_bf16<<<dim3((unsigned)(MD / 1024)), cb, 0, stream>>>(q, Qb);
  cvt_f32_bf16<<<dim3((unsigned)(MD / 1024)), cb, 0, stream>>>(k, Kb);
  cvt_f32_bf16<<<dim3((unsigned)(MD / 1024)), cb, 0, stream>>>(v, Vb);
  cvt_f32_bf16<<<dim3((unsigned)(DD / 1024)), cb, 0, stream>>>(w_q, Wqb);
  cvt_f32_bf16<<<dim3((unsigned)(DD / 1024)), cb, 0, stream>>>(w_k, Wkb);
  cvt_f32_bf16<<<dim3((unsigned)(DD / 1024)), cb, 0, stream>>>(w_v, Wvb);
  cvt_f32_bf16<<<dim3((unsigned)(DD / 1024)), cb, 0, stream>>>(w_o, Wob);

  dim3 grid(N / TN, M / TM);         // 8 x 64
  dim3 block(256);
  gemm_bias_wmma<bf16><<<grid, block, 0, stream>>>(Qb, Wqb, b_q, Qlin, M, N, K);
  gemm_bias_wmma<bf16><<<grid, block, 0, stream>>>(Kb, Wkb, b_k, Klin, M, N, K);
  gemm_bias_wmma<bf16><<<grid, block, 0, stream>>>(Vb, Wvb, b_v, Vlin, M, N, K);

  token_attention<<<dim3(M), dim3(64), 0, stream>>>(Qlin, Klin, Vlin, Qlin);

  gemm_bias_wmma<float><<<grid, block, 0, stream>>>(Qlin, Wob, b_o, out, M, N, K);
}
